// HL_HGCNN_pepfunc_dense_int3_attpool_4269197492231
// MI455X (gfx1250) — matrix operand order repaired and checked by reading, not compiled, not run on hardware
//
#include <hip/hip_runtime.h>
#include <hip/hip_bf16.h>

// ---------------------------------------------------------------------------
// Types for CDNA5 WMMA (gfx1250, wave32)
// ---------------------------------------------------------------------------
typedef __attribute__((ext_vector_type(16))) __bf16 v16bf;
typedef __attribute__((ext_vector_type(8)))  float  v8f;

struct U4 { unsigned x, y, z, w; };
union Frag { v16bf bf; U4 q[2]; };

__device__ __forceinline__ unsigned short f2bf(float f) {
  unsigned u = __float_as_uint(f);
  u += 0x7fffu + ((u >> 16) & 1u);         // round to nearest even
  return (unsigned short)(u >> 16);
}

// ---------------------------------------------------------------------------
// WMMA GEMM:  C[M,N] = epi( concat(A1[:, :K1], A2)  @ B[K,N] )
//   A1 may have an elementwise subtrahend Asub (same lda) -> (A1 - Asub)
//   epi: (+bias[n]) (+C_old if beta) (*rowscale[m])
// Block = 256 threads = 8 wave32 waves; tile 128x64; each wave owns a 32x32
// C slab -> 2 A-frags x 2 B-frags = 4 WMMAs per K-step with operand reuse.
// ---------------------------------------------------------------------------
#define BM 128
#define BN 64
#define TK 32

__device__ __forceinline__ void epi_store(const v8f& acc, int gm0, int gn, int khalf,
                                          int M, int N,
                                          const float* __restrict__ bias,
                                          const float* __restrict__ rowscale,
                                          float* __restrict__ C, long long ldc, int beta) {
  if (gn >= N) return;
  #pragma unroll
  for (int v = 0; v < 8; ++v) {
    int gm = gm0 + khalf * 8 + v;          // C layout: VGPR v -> M = v + 8*khalf
    if (gm >= M) continue;
    float o = acc[v];
    if (bias) o += bias[gn];
    if (beta) o += C[(long long)gm * ldc + gn];
    if (rowscale) o *= rowscale[gm];
    C[(long long)gm * ldc + gn] = o;
  }
}

__global__ __launch_bounds__(256)
void gemm_wmma(const float* __restrict__ A1, long long lda1, int K1,
               const float* __restrict__ A2, long long lda2,
               const float* __restrict__ Asub,
               const float* __restrict__ Bw,        // K x N row-major
               const float* __restrict__ bias,      // or null
               const float* __restrict__ rowscale,  // or null
               float* __restrict__ C, long long ldc,
               int M, int N, int K, int beta)
{
  __shared__ unsigned short As[BM][TK + 8];   // 128 x 40 ushort, 16B-aligned rows
  __shared__ unsigned short Bs[BN][TK + 8];   // [n][k] (K-transposed)

  const int bm   = blockIdx.x * BM;
  const int bn   = blockIdx.y * BN;
  const int tid  = threadIdx.x;
  const int lane = tid & 31;
  const int wave = tid >> 5;
  const int wm   = (wave >> 1) * 32;   // 0,32,64,96
  const int wn   = (wave & 1) * 32;    // 0,32

  v8f acc00 = {}, acc01 = {}, acc10 = {}, acc11 = {};

  for (int kt = 0; kt < K; kt += TK) {
    // ---- stage A tile (128 x 32) as bf16, with virtual concat + subtract ----
    #pragma unroll
    for (int i = 0; i < (BM * TK) / 256; ++i) {
      int idx = tid + i * 256;
      int r = idx >> 5, k = idx & 31;
      int gr = bm + r, gk = kt + k;
      float v = 0.f;
      if (gr < M && gk < K) {
        if (gk < K1) {
          v = A1[(long long)gr * lda1 + gk];
          if (Asub) v -= Asub[(long long)gr * lda1 + gk];
        } else {
          v = A2[(long long)gr * lda2 + (gk - K1)];
        }
      }
      As[r][k] = f2bf(v);
    }
    // ---- stage B tile (32 x 64) K-transposed ----
    #pragma unroll
    for (int i = 0; i < (TK * BN) / 256; ++i) {
      int idx = tid + i * 256;
      int n = idx & 63, k = idx >> 6;
      int gk = kt + k, gn = bn + n;
      float v = 0.f;
      if (gk < K && gn < N) v = Bw[(long long)gk * N + gn];
      Bs[n][k] = f2bf(v);
    }
    // prefetch next A tile rows (speculative; emits global_prefetch)
    if (kt + TK < K1) {
      __builtin_prefetch(&A1[(long long)(bm + (tid & 127)) * lda1 + kt + TK], 0, 1);
    }
    __syncthreads();

    // ---- fragments per ISA layout ----
    const int khalf = lane >> 4;             // K half-select
    const int nl    = lane & 15;
    const int am0   = wm + nl;               // A rows for frag 0
    const int am1   = wm + 16 + nl;          // A rows for frag 1
    Frag fa0, fa1;
    fa0.q[0] = *reinterpret_cast<const U4*>(&As[am0][khalf * 8]);       // K 0..7 / 8..15
    fa0.q[1] = *reinterpret_cast<const U4*>(&As[am0][16 + khalf * 8]);  // K 16..23 / 24..31
    fa1.q[0] = *reinterpret_cast<const U4*>(&As[am1][khalf * 8]);
    fa1.q[1] = *reinterpret_cast<const U4*>(&As[am1][16 + khalf * 8]);

    const int kb = khalf * 8;                // lanes 0-15: K0-7,16-23; lanes 16-31: K8-15,24-31
    Frag fb0, fb1;
    fb0.q[0] = *reinterpret_cast<const U4*>(&Bs[wn + nl][kb]);
    fb0.q[1] = *reinterpret_cast<const U4*>(&Bs[wn + nl][kb + 16]);
    fb1.q[0] = *reinterpret_cast<const U4*>(&Bs[wn + 16 + nl][kb]);
    fb1.q[1] = *reinterpret_cast<const U4*>(&Bs[wn + 16 + nl][kb + 16]);

    acc00 = __builtin_amdgcn_wmma_f32_16x16x32_bf16(false, fa0.bf, false, fb0.bf,
                                                    (short)0, acc00, false, false);
    acc01 = __builtin_amdgcn_wmma_f32_16x16x32_bf16(false, fa0.bf, false, fb1.bf,
                                                    (short)0, acc01, false, false);
    acc10 = __builtin_amdgcn_wmma_f32_16x16x32_bf16(false, fa1.bf, false, fb0.bf,
                                                    (short)0, acc10, false, false);
    acc11 = __builtin_amdgcn_wmma_f32_16x16x32_bf16(false, fa1.bf, false, fb1.bf,
                                                    (short)0, acc11, false, false);
    __syncthreads();
  }

  // ---- epilogue ----
  const int khalf = lane >> 4;
  const int nl    = lane & 15;
  const int gn0   = bn + wn + nl;
  const int gn1   = gn0 + 16;
  epi_store(acc00, bm + wm,      gn0, khalf, M, N, bias, rowscale, C, ldc, beta);
  epi_store(acc01, bm + wm,      gn1, khalf, M, N, bias, rowscale, C, ldc, beta);
  epi_store(acc10, bm + wm + 16, gn0, khalf, M, N, bias, rowscale, C, ldc, beta);
  epi_store(acc11, bm + wm + 16, gn1, khalf, M, N, bias, rowscale, C, ldc, beta);
}

// ---------------------------------------------------------------------------
// Elementwise / scatter / reduction kernels
// ---------------------------------------------------------------------------
__global__ void zero_f32(float* p, long long n) {
  long long i = (long long)blockIdx.x * blockDim.x + threadIdx.x;
  if (i < n) p[i] = 0.f;
}

__global__ void add_scalar(float* p, float v, long long n) {
  long long i = (long long)blockIdx.x * blockDim.x + threadIdx.x;
  if (i < n) p[i] += v;
}

// agg[node] += x_s[edge] scattered to both endpoints
__global__ void scatter_edge_sum2(float* __restrict__ agg, int dim,
                                  const int* __restrict__ src, const int* __restrict__ dst,
                                  const float* __restrict__ X, long long ldx, long long n) {
  long long i = (long long)blockIdx.x * blockDim.x + threadIdx.x;
  if (i >= n) return;
  long long e = i / dim; int k = (int)(i % dim);
  float v = X[e * ldx + k];
  atomicAdd(&agg[(long long)src[e] * dim + k], v);
  atomicAdd(&agg[(long long)dst[e] * dim + k], v);
}

__global__ void rows_mul_recip(float* __restrict__ A, int dim,
                               const float* __restrict__ D, long long n) {
  long long i = (long long)blockIdx.x * blockDim.x + threadIdx.x;
  if (i >= n) return;
  long long r = i / dim;
  A[i] = A[i] / D[r];
}

__global__ void gather_edge_mean(float* __restrict__ agg, int dim,
                                 const int* __restrict__ src, const int* __restrict__ dst,
                                 const float* __restrict__ X, long long ldx, long long n) {
  long long i = (long long)blockIdx.x * blockDim.x + threadIdx.x;
  if (i >= n) return;
  long long e = i / dim; int k = (int)(i % dim);
  agg[i] = 0.5f * (X[(long long)src[e] * ldx + k] + X[(long long)dst[e] * ldx + k]);
}

// Lx[row[e]] += ew[e] * X[col[e]]
__global__ void scatter_L(float* __restrict__ Lx, int f,
                          const int* __restrict__ row, const int* __restrict__ col,
                          const float* __restrict__ ew,
                          const float* __restrict__ X, long long n) {
  long long i = (long long)blockIdx.x * blockDim.x + threadIdx.x;
  if (i >= n) return;
  long long e = i / f; int k = (int)(i % f);
  atomicAdd(&Lx[(long long)row[e] * f + k], ew[e] * X[(long long)col[e] * f + k]);
}

// att[r] = (1-l) + l * sigmoid( dot(concat(X[r], AGG[r]), W) + b )
__global__ void rowdot_att(const float* __restrict__ X, long long ldx, int K1,
                           const float* __restrict__ AGG, long long ldagg, int K,
                           const float* __restrict__ W, const float* __restrict__ b,
                           float l, float* __restrict__ att, int M) {
  int row  = blockIdx.x * (blockDim.x >> 5) + (threadIdx.x >> 5);
  int lane = threadIdx.x & 31;
  if (row >= M) return;
  float s = 0.f;
  for (int k = lane; k < K1; k += 32) s += X[(long long)row * ldx + k] * W[k];
  for (int k = K1 + lane; k < K; k += 32) s += AGG[(long long)row * ldagg + (k - K1)] * W[k];
  #pragma unroll
  for (int off = 16; off; off >>= 1) s += __shfl_down(s, off, 32);
  if (lane == 0) {
    float sig = 1.f / (1.f + __expf(-(s + b[0])));
    att[row] = (1.f - l) + l * sig;
  }
}

__global__ void bn_stats(const float* __restrict__ Y, int ldy, int M,
                         float* __restrict__ mean, float* __restrict__ var) {
  __shared__ float sh[256], sh2[256];
  int c = blockIdx.x;
  float s = 0.f, s2 = 0.f;
  for (int r = threadIdx.x; r < M; r += blockDim.x) {
    float v = Y[(long long)r * ldy + c];
    s += v; s2 += v * v;
  }
  sh[threadIdx.x] = s; sh2[threadIdx.x] = s2;
  __syncthreads();
  for (int o = 128; o; o >>= 1) {
    if ((int)threadIdx.x < o) { sh[threadIdx.x] += sh[threadIdx.x + o]; sh2[threadIdx.x] += sh2[threadIdx.x + o]; }
    __syncthreads();
  }
  if (threadIdx.x == 0) {
    float m = sh[0] / (float)M;
    mean[c] = m;
    var[c]  = sh2[0] / (float)M - m * m;
  }
}

// Y <- relu(bn(Y)); also written into X0[:, coloff:coloff+f] (the concat)
__global__ void bn_apply_relu(float* __restrict__ Y, int f,
                              const float* __restrict__ mean, const float* __restrict__ var,
                              const float* __restrict__ g, const float* __restrict__ be,
                              float* __restrict__ X0, long long ld0, int coloff, long long n) {
  long long i = (long long)blockIdx.x * blockDim.x + threadIdx.x;
  if (i >= n) return;
  long long r = i / f; int c = (int)(i % f);
  float v = (Y[i] - mean[c]) * rsqrtf(var[c] + 1e-5f) * g[c] + be[c];
  v = fmaxf(v, 0.f);
  Y[i] = v;
  X0[r * ld0 + coloff + c] = v;
}

__global__ void scale_rows(float* __restrict__ X, long long ld, int dim,
                           const float* __restrict__ s, long long n) {
  long long i = (long long)blockIdx.x * blockDim.x + threadIdx.x;
  if (i >= n) return;
  long long r = i / dim; int k = (int)(i % dim);
  X[r * ld + k] *= s[r];
}

__global__ void count_segments(float* __restrict__ cnt, const int* __restrict__ pos, long long n) {
  long long i = (long long)blockIdx.x * blockDim.x + threadIdx.x;
  if (i < n) atomicAdd(&cnt[pos[i]], 1.f);
}

__global__ void degree_count(float* __restrict__ D, const int* __restrict__ src,
                             const int* __restrict__ dst, long long n) {
  long long i = (long long)blockIdx.x * blockDim.x + threadIdx.x;
  if (i >= n) return;
  atomicAdd(&D[src[i]], 1.f);
  atomicAdd(&D[dst[i]], 1.f);
}

__global__ void scatter_rows_add(float* __restrict__ post, long long ldp,
                                 const int* __restrict__ pos,
                                 const float* __restrict__ pre, long long ldpre,
                                 int dim, long long n) {
  long long i = (long long)blockIdx.x * blockDim.x + threadIdx.x;
  if (i >= n) return;
  long long r = i / dim; int k = (int)(i % dim);
  atomicAdd(&post[(long long)pos[r] * ldp + k], pre[r * ldpre + k]);
}

__global__ void rows_div_count(float* __restrict__ X, long long ld, int dim,
                               const float* __restrict__ cnt, long long n) {
  long long i = (long long)blockIdx.x * blockDim.x + threadIdx.x;
  if (i >= n) return;
  long long r = i / dim; int k = (int)(i % dim);
  X[r * ld + k] /= fmaxf(cnt[r], 1.f);
}

// out[128,10] = concat(xs,xt)[128,1024] @ W[1024,10] + b
__global__ void final_linear(const float* __restrict__ xs, const float* __restrict__ xt,
                             const float* __restrict__ W, const float* __restrict__ b,
                             float* __restrict__ out) {
  int idx = blockIdx.x * blockDim.x + threadIdx.x;
  if (idx >= 128 * 10) return;
  int r = idx / 10, c = idx % 10;
  float s = b[c];
  for (int k = 0; k < 512; ++k) s += xs[r * 512 + k] * W[k * 10 + c];
  for (int k = 0; k < 512; ++k) s += xt[r * 512 + k] * W[(512 + k) * 10 + c];
  out[idx] = s;
}

// ---------------------------------------------------------------------------
// Orchestration
// ---------------------------------------------------------------------------
extern "C" void kernel_launch(void* const* d_in, const int* in_sizes, int n_in,
                              void* d_out, int out_size, void* d_ws, size_t ws_size,
                              hipStream_t stream) {
  (void)in_sizes; (void)n_in; (void)out_size; (void)ws_size;

  const int N0 = 64000, E0 = 128000, N1 = 16000, E1 = 32000, BB = 128;
  const int FIL[4] = {64, 128, 256, 512};

  const float* x_t_feat = (const float*)d_in[0];
  const float* x_s_feat = (const float*)d_in[1];
  const int*   pos_t    = (const int*)d_in[2];
  const int*   pos_s    = (const int*)d_in[3];
  const int*   ei0      = (const int*)d_in[4];
  const int*   ei1      = (const int*)d_in[5];
  const int*   eit0     = (const int*)d_in[6];  const float* ewt0 = (const float*)d_in[7];
  const int*   eis0     = (const int*)d_in[8];  const float* ews0 = (const float*)d_in[9];
  const int*   eit1     = (const int*)d_in[10]; const float* ewt1 = (const float*)d_in[11];
  const int*   eis1     = (const int*)d_in[12]; const float* ews1 = (const float*)d_in[13];
  const int*   n_batch1 = (const int*)d_in[14];
  const int*   s_batch1 = (const int*)d_in[15];
  const float* const* P = (const float* const*)(d_in + 16);
  // P: 0..7 init params; per (i,j): base 8+40i+18j -> Wt,bt,Ws,bs,atW,atb,asW,asb,
  //     tW0,tW1,tb,tg,tbe,sW0,sW1,sb,sg,sbe ; att(i): 8+40i+36 -> tW,tb,sW,sb ; 168,169 out

  // ---- workspace carve (floats) ----
  float* w = (float*)d_ws;
  size_t off = 0;
  auto take = [&](size_t n) { float* p = w + off; off += n; return p; };
  float* XT0p = take((size_t)N0 * 192);
  float* XS0p = take((size_t)E0 * 192);
  float* XT0q = take((size_t)N1 * 1984);
  float* XS0q = take((size_t)E1 * 1984);
  float* AGGT = take(32000000);
  float* AGGS = take(64000000);
  float* XTne = take(8192000);
  float* LXT  = take(8192000);
  float* YT   = take(8192000);
  float* XSne = take(16384000);
  float* LXS  = take(16384000);
  float* YS   = take(16384000);
  float* Dg   = take(64000);
  float* ATTt = take(64000);
  float* ATTs = take(128000);
  float* MEAN = take(512);
  float* VARb = take(512);
  float* CNT  = take(32000);
  float* XSM  = take((size_t)BB * 512);
  float* XTM  = take((size_t)BB * 512);
  float* CNTB = take(BB);

  dim3 blk(256);
  auto G1 = [](long long n) { return dim3((unsigned)((n + 255) / 256)); };
  auto GG = [](int M, int N) { return dim3((unsigned)((M + BM - 1) / BM),
                                           (unsigned)((N + BN - 1) / BN)); };

  // ---- init layers: x = bn_relu(feat @ W + b) ----
  gemm_wmma<<<GG(N0, 64), blk, 0, stream>>>(x_t_feat, 29, 29, nullptr, 0, nullptr,
      P[0], P[1], nullptr, YT, 64, N0, 64, 29, 0);
  bn_stats<<<dim3(64), blk, 0, stream>>>(YT, 64, N0, MEAN, VARb);
  bn_apply_relu<<<G1((long long)N0 * 64), blk, 0, stream>>>(YT, 64, MEAN, VARb, P[2], P[3],
      XT0p, 192, 0, (long long)N0 * 64);

  gemm_wmma<<<GG(E0, 64), blk, 0, stream>>>(x_s_feat, 23, 23, nullptr, 0, nullptr,
      P[4], P[5], nullptr, YS, 64, E0, 64, 23, 0);
  bn_stats<<<dim3(64), blk, 0, stream>>>(YS, 64, E0, MEAN, VARb);
  bn_apply_relu<<<G1((long long)E0 * 64), blk, 0, stream>>>(YS, 64, MEAN, VARb, P[6], P[7],
      XS0p, 192, 0, (long long)E0 * 64);

  // ---- level state ----
  int Mt = N0, Ms = E0;
  float *X0t = XT0p, *X0s = XS0p;
  long long ldt = 192, lds = 192;
  const int *src = ei0, *dst = ei0 + E0;
  const int *etI = eit0; const float* etW = ewt0; long long etN = N0 + 2LL * E0;
  const int *esI = eis0; const float* esW = ews0; long long esN = 4LL * E0;

  zero_f32<<<G1(Mt), blk, 0, stream>>>(Dg, Mt);
  degree_count<<<G1(Ms), blk, 0, stream>>>(Dg, src, dst, Ms);
  add_scalar<<<G1(Mt), blk, 0, stream>>>(Dg, 1e-6f, Mt);

  int dim = 64;

  auto make_agg = [&](int d) {
    zero_f32<<<G1((long long)Mt * d), blk, 0, stream>>>(AGGT, (long long)Mt * d);
    scatter_edge_sum2<<<G1((long long)Ms * d), blk, 0, stream>>>(AGGT, d, src, dst, X0s, lds,
        (long long)Ms * d);
    rows_mul_recip<<<G1((long long)Mt * d), blk, 0, stream>>>(AGGT, d, Dg, (long long)Mt * d);
    gather_edge_mean<<<G1((long long)Ms * d), blk, 0, stream>>>(AGGS, d, src, dst, X0t, ldt,
        (long long)Ms * d);
  };

  for (int i = 0; i < 4; ++i) {
    int f = FIL[i];
    for (int j = 0; j < 2; ++j) {
      const int pb = 8 + 40 * i + 18 * j;
      const float *Wt = P[pb+0], *bt = P[pb+1], *Ws = P[pb+2], *bs = P[pb+3];
      const float *atW = P[pb+4], *atb = P[pb+5], *asW = P[pb+6], *asb = P[pb+7];
      const float *tW0 = P[pb+8], *tW1 = P[pb+9], *tb = P[pb+10], *tg = P[pb+11], *tbe = P[pb+12];
      const float *sW0 = P[pb+13], *sW1 = P[pb+14], *sb = P[pb+15], *sg = P[pb+16], *sbe = P[pb+17];

      // ne_int: agg, attention scalars, fused concat-GEMM with row scaling
      make_agg(dim);
      rowdot_att<<<dim3((Mt + 7) / 8), blk, 0, stream>>>(X0t, ldt, dim, AGGT, dim, 2 * dim,
          atW, atb, 1.0f, ATTt, Mt);
      rowdot_att<<<dim3((Ms + 7) / 8), blk, 0, stream>>>(X0s, lds, dim, AGGS, dim, 2 * dim,
          asW, asb, 1.0f, ATTs, Ms);
      gemm_wmma<<<GG(Mt, f), blk, 0, stream>>>(X0t, ldt, dim, AGGT, dim, nullptr,
          Wt, bt, ATTt, XTne, f, Mt, f, 2 * dim, 0);
      gemm_wmma<<<GG(Ms, f), blk, 0, stream>>>(X0s, lds, dim, AGGS, dim, nullptr,
          Ws, bs, ATTs, XSne, f, Ms, f, 2 * dim, 0);

      // hl2 (t): Y = x@W0 + b ; Y += (x - Lx)@W1 ; then BN+ReLU -> concat into X0
      zero_f32<<<G1((long long)Mt * f), blk, 0, stream>>>(LXT, (long long)Mt * f);
      scatter_L<<<G1(etN * f), blk, 0, stream>>>(LXT, f, etI, etI + etN, etW, XTne, etN * f);
      gemm_wmma<<<GG(Mt, f), blk, 0, stream>>>(XTne, f, f, nullptr, 0, nullptr,
          tW0, tb, nullptr, YT, f, Mt, f, f, 0);
      gemm_wmma<<<GG(Mt, f), blk, 0, stream>>>(XTne, f, f, nullptr, 0, LXT,
          tW1, nullptr, nullptr, YT, f, Mt, f, f, 1);
      bn_stats<<<dim3(f), blk, 0, stream>>>(YT, f, Mt, MEAN, VARb);
      bn_apply_relu<<<G1((long long)Mt * f), blk, 0, stream>>>(YT, f, MEAN, VARb, tg, tbe,
          X0t, ldt, dim, (long long)Mt * f);

      // hl2 (s)
      zero_f32<<<G1((long long)Ms * f), blk, 0, stream>>>(LXS, (long long)Ms * f);
      scatter_L<<<G1(esN * f), blk, 0, stream>>>(LXS, f, esI, esI + esN, esW, XSne, esN * f);
      gemm_wmma<<<GG(Ms, f), blk, 0, stream>>>(XSne, f, f, nullptr, 0, nullptr,
          sW0, sb, nullptr, YS, f, Ms, f, f, 0);
      gemm_wmma<<<GG(Ms, f), blk, 0, stream>>>(XSne, f, f, nullptr, 0, LXS,
          sW1, nullptr, nullptr, YS, f, Ms, f, f, 1);
      bn_stats<<<dim3(f), blk, 0, stream>>>(YS, f, Ms, MEAN, VARb);
      bn_apply_relu<<<G1((long long)Ms * f), blk, 0, stream>>>(YS, f, MEAN, VARb, sg, sbe,
          X0s, lds, dim, (long long)Ms * f);

      dim += f;
    }

    // attention block: x0 *= (0.5 + 0.5*sigmoid(z @ W + b))
    {
      const int pa = 8 + 40 * i + 36;
      int d = dim;
      make_agg(d);
      rowdot_att<<<dim3((Mt + 7) / 8), blk, 0, stream>>>(X0t, ldt, d, AGGT, d, 2 * d,
          P[pa+0], P[pa+1], 0.5f, ATTt, Mt);
      rowdot_att<<<dim3((Ms + 7) / 8), blk, 0, stream>>>(X0s, lds, d, AGGS, d, 2 * d,
          P[pa+2], P[pa+3], 0.5f, ATTs, Ms);
      scale_rows<<<G1((long long)Mt * d), blk, 0, stream>>>(X0t, ldt, d, ATTt, (long long)Mt * d);
      scale_rows<<<G1((long long)Ms * d), blk, 0, stream>>>(X0s, lds, d, ATTs, (long long)Ms * d);
    }

    // hierarchical scatter-mean pooling after block 0
    if (i == 0) {
      zero_f32<<<G1((long long)N1 * 1984), blk, 0, stream>>>(XT0q, (long long)N1 * 1984);
      zero_f32<<<G1(N1), blk, 0, stream>>>(CNT, N1);
      scatter_rows_add<<<G1((long long)N0 * 192), blk, 0, stream>>>(XT0q, 1984, pos_t,
          XT0p, 192, 192, (long long)N0 * 192);
      count_segments<<<G1(N0), blk, 0, stream>>>(CNT, pos_t, N0);
      rows_div_count<<<G1((long long)N1 * 192), blk, 0, stream>>>(XT0q, 1984, 192, CNT,
          (long long)N1 * 192);

      zero_f32<<<G1((long long)E1 * 1984), blk, 0, stream>>>(XS0q, (long long)E1 * 1984);
      zero_f32<<<G1(E1), blk, 0, stream>>>(CNT, E1);
      scatter_rows_add<<<G1((long long)E0 * 192), blk, 0, stream>>>(XS0q, 1984, pos_s,
          XS0p, 192, 192, (long long)E0 * 192);
      count_segments<<<G1(E0), blk, 0, stream>>>(CNT, pos_s, E0);
      rows_div_count<<<G1((long long)E1 * 192), blk, 0, stream>>>(XS0q, 1984, 192, CNT,
          (long long)E1 * 192);

      Mt = N1; Ms = E1; X0t = XT0q; X0s = XS0q; ldt = 1984; lds = 1984;
      src = ei1; dst = ei1 + E1;
      etI = eit1; etW = ewt1; etN = N1 + 2LL * E1;
      esI = eis1; esW = ews1; esN = 4LL * E1;
      zero_f32<<<G1(Mt), blk, 0, stream>>>(Dg, Mt);
      degree_count<<<G1(Ms), blk, 0, stream>>>(Dg, src, dst, Ms);
      add_scalar<<<G1(Mt), blk, 0, stream>>>(Dg, 1e-6f, Mt);
    }
  }

  // ---- readout: batch seg-means of last conv outputs (YS, YT), then linear ----
  zero_f32<<<G1((long long)BB * 512), blk, 0, stream>>>(XSM, (long long)BB * 512);
  zero_f32<<<G1(BB), blk, 0, stream>>>(CNTB, BB);
  scatter_rows_add<<<G1((long long)E1 * 512), blk, 0, stream>>>(XSM, 512, s_batch1,
      YS, 512, 512, (long long)E1 * 512);
  count_segments<<<G1(E1), blk, 0, stream>>>(CNTB, s_batch1, E1);
  rows_div_count<<<G1((long long)BB * 512), blk, 0, stream>>>(XSM, 512, 512, CNTB,
      (long long)BB * 512);

  zero_f32<<<G1((long long)BB * 512), blk, 0, stream>>>(XTM, (long long)BB * 512);
  zero_f32<<<G1(BB), blk, 0, stream>>>(CNTB, BB);
  scatter_rows_add<<<G1((long long)N1 * 512), blk, 0, stream>>>(XTM, 512, n_batch1,
      YT, 512, 512, (long long)N1 * 512);
  count_segments<<<G1(N1), blk, 0, stream>>>(CNTB, n_batch1, N1);
  rows_div_count<<<G1((long long)BB * 512), blk, 0, stream>>>(XTM, 512, 512, CNTB,
      (long long)BB * 512);

  final_linear<<<dim3(5), blk, 0, stream>>>(XSM, XTM, P[168], P[169], (float*)d_out);
}
